// GraphTransformerLayer_30348238914117
// MI455X (gfx1250) — matrix operand order repaired and checked by reading, not compiled, not run on hardware
//
#include <hip/hip_runtime.h>
#include <hip/hip_bf16.h>
#include <math.h>

// ---------------------------------------------------------------------------
// Types for gfx1250 WMMA / TDM
// ---------------------------------------------------------------------------
typedef __attribute__((ext_vector_type(16))) __bf16 bf16x16;
typedef __attribute__((ext_vector_type(8)))  __bf16 bf16x8;
typedef __attribute__((ext_vector_type(8)))  float  f32x8;
typedef __attribute__((ext_vector_type(4)))  float  f32x4;
typedef __attribute__((ext_vector_type(4)))  unsigned int u32x4;
typedef __attribute__((ext_vector_type(8)))  int i32x8;
typedef __attribute__((ext_vector_type(4)))  int i32x4;

#define N_NODES 65536
#define DMODEL  1024
#define NHEAD   16
#define DHEAD   64
#define FFDIM   2048
#define BN_EPS  1e-5f

// ---------------------------------------------------------------------------
// fp32 -> bf16 elementwise convert (for h)
// ---------------------------------------------------------------------------
__global__ __launch_bounds__(256)
void k_f32_to_bf16(const float* __restrict__ x, __bf16* __restrict__ y, long n) {
  long i = ((long)blockIdx.x * 256 + threadIdx.x) * 4;
  if (i + 3 < n) {
    f32x4 v = *(const f32x4*)(x + i);
    y[i + 0] = (__bf16)v.x;
    y[i + 1] = (__bf16)v.y;
    y[i + 2] = (__bf16)v.z;
    y[i + 3] = (__bf16)v.w;
  }
}

// ---------------------------------------------------------------------------
// Transpose + convert weights: W[K,N] fp32 -> Wt[N,K] bf16
// ---------------------------------------------------------------------------
__global__ __launch_bounds__(256)
void k_transpose_bf16(const float* __restrict__ W, __bf16* __restrict__ Wt,
                      int K, int N) {
  long i = (long)blockIdx.x * 256 + threadIdx.x;
  if (i < (long)K * N) {
    int k = (int)(i / N);
    int n = (int)(i % N);
    Wt[(long)n * K + k] = (__bf16)W[i];
  }
}

// ---------------------------------------------------------------------------
// TDM: async-DMA one 128-row x 64-half tile (row stride K halves) into LDS,
// padding 8 halves after every 64 (pitch 72) to keep WMMA fragment reads
// bank-conflict-free.  D# per CDNA5 ISA ch.8.
// ---------------------------------------------------------------------------
__device__ __forceinline__
void tdm_load_tile(unsigned lds_byte, const __bf16* gptr, int K) {
  unsigned long long ga = (unsigned long long)(uintptr_t)gptr;
  u32x4 g0;
  g0.x = 1u;                                           // count=1 (valid D#)
  g0.y = lds_byte;                                     // lds_addr (bytes)
  g0.z = (unsigned)(ga & 0xffffffffull);               // global_addr[31:0]
  g0.w = (unsigned)((ga >> 32) & 0x01ffffffull)        // global_addr[56:32]
         | (2u << 30);                                 // type=2 ("image")
  i32x8 g1;
  g1[0] = (1 << 16)        // data_size = 2 bytes
        | (1 << 20)        // pad_enable
        | (4 << 22)        // pad_interval: code 4 -> every 32 DWORDs (128 B)
        | (3 << 25);       // pad_amount:   code 3 -> 4 DWORDs (16 B)
  g1[1] = (K & 0xffff) << 16;       // tensor_dim0[15:0] = K
  g1[2] = 0;                        // tensor_dim0 hi, tensor_dim1 lo
  g1[3] = 16 | (64 << 16);          // tensor_dim1 = 1<<20, tile_dim0 = 64
  g1[4] = 128;                      // tile_dim1 = 128 rows, tile_dim2 = 0
  g1[5] = K;                        // tensor_dim0_stride = K elements
  g1[6] = 0;
  g1[7] = 0;
  i32x4 z4 = {0, 0, 0, 0};                    // groups 2/3 unused (2-D tile)
  i32x8 z8 = {0, 0, 0, 0, 0, 0, 0, 0};        // trailing group (clang-23 form)
  __builtin_amdgcn_tensor_load_to_lds(g0, g1, z4, z4, z8, 0);
}

// ---------------------------------------------------------------------------
// bf16 WMMA GEMM: C[M,N] = A[M,K] @ B[K,N]   (B supplied transposed: Bt[N,K])
// Tile 128x128x64, 256 threads = 8 waves (2x4), each wave 64x32 = 4x2 WMMA.
// Tiles fed by TDM (tensor_load_to_lds), double-buffered on TENSORcnt.
// mode 0: store bf16
// mode 1: store fp32 = acc + bias[col] + resid[row,col]
// mode 2: store bf16 = relu(acc + bias[col])
// ---------------------------------------------------------------------------
#define LDSP 72          // halves; row*36 mod 64 is a 16-row permutation
#define TILE_HALVES (128 * LDSP)          // 9216 halves per tile
#define STAGE_HALVES (2 * TILE_HALVES)    // A tile + B tile per stage

__global__ __launch_bounds__(256)
void k_gemm_bf16(const __bf16* __restrict__ A, const __bf16* __restrict__ Bt,
                 int M, int N, int K, int mode,
                 const float* __restrict__ bias,
                 const float* __restrict__ resid,
                 float* __restrict__ outF,
                 __bf16* __restrict__ outB) {
  __shared__ __bf16 smem[2 * STAGE_HALVES];   // 72 KB, double-buffered

  const int tid  = threadIdx.x;
  const int lane = tid & 31;
  const int wave = tid >> 5;
  const int wm   = wave >> 2;   // 0..1
  const int wn   = wave & 3;    // 0..3
  const long r0  = (long)blockIdx.x * 128;
  const long c0  = (long)blockIdx.y * 128;

  const unsigned smem_base = (unsigned)(uintptr_t)&smem[0]; // LDS byte offset
  const __bf16* Abase = A + r0 * (long)K;
  const __bf16* Bbase = Bt + c0 * (long)K;

  f32x8 acc[4][2] = {};

  // prologue: stage 0
  if (wave == 0) {
    tdm_load_tile(smem_base, Abase, K);
    tdm_load_tile(smem_base + TILE_HALVES * 2, Bbase, K);
  }

  const int nIter = K >> 6;           // K/64
  for (int it = 0; it < nIter; ++it) {
    const int k0  = it << 6;
    const int cur = it & 1;
    if (wave == 0) {
      if (it + 1 < nIter) {
        unsigned nb = smem_base + (unsigned)((1 - cur) * STAGE_HALVES * 2);
        tdm_load_tile(nb, Abase + (k0 + 64), K);
        tdm_load_tile(nb + TILE_HALVES * 2, Bbase + (k0 + 64), K);
        __builtin_amdgcn_s_wait_tensorcnt(2);  // current stage complete
      } else {
        __builtin_amdgcn_s_wait_tensorcnt(0);
      }
    }
    __syncthreads();

    const __bf16* As = &smem[cur * STAGE_HALVES];
    const __bf16* Bs = As + TILE_HALVES;

#pragma unroll
    for (int kk = 0; kk < 64; kk += 32) {
      // A frag: lanes 0-15 row m hold K {kk..kk+7, kk+16..kk+23};
      //         lanes 16-31 hold K {kk+8..kk+15, kk+24..kk+31}
      const int akb = kk + ((lane & 16) ? 8 : 0);
      bf16x16 af[4];
#pragma unroll
      for (int mt = 0; mt < 4; ++mt) {
        int row = wm * 64 + mt * 16 + (lane & 15);
        union { bf16x16 v; bf16x8 h[2]; } u;
        u.h[0] = *(const bf16x8*)&As[row * LDSP + akb];
        u.h[1] = *(const bf16x8*)&As[row * LDSP + akb + 16];
        af[mt] = u.v;
      }
      // B frag: lanes 0-15 col n hold K {kk..kk+7, kk+16..kk+23};
      //         lanes 16-31 hold K {kk+8..kk+15, kk+24..kk+31}
      const int bkb = kk + ((lane & 16) ? 8 : 0);
      bf16x16 bfr[2];
#pragma unroll
      for (int nt = 0; nt < 2; ++nt) {
        int col = wn * 32 + nt * 16 + (lane & 15);
        union { bf16x16 v; bf16x8 h[2]; } u;
        u.h[0] = *(const bf16x8*)&Bs[col * LDSP + bkb];
        u.h[1] = *(const bf16x8*)&Bs[col * LDSP + bkb + 16];
        bfr[nt] = u.v;
      }
#pragma unroll
      for (int mt = 0; mt < 4; ++mt)
#pragma unroll
        for (int nt = 0; nt < 2; ++nt)
          acc[mt][nt] = __builtin_amdgcn_wmma_f32_16x16x32_bf16(
              false, af[mt], false, bfr[nt], (short)0, acc[mt][nt],
              false, false);
    }
    __syncthreads();   // all waves done with buf[cur] before TDM overwrites it
  }

  // Epilogue. C layout: VGPR j -> row (j | (lane&16 ? 8:0)), col = lane&15.
#pragma unroll
  for (int mt = 0; mt < 4; ++mt)
#pragma unroll
    for (int nt = 0; nt < 2; ++nt) {
      long rowb = r0 + wm * 64 + mt * 16 + ((lane & 16) ? 8 : 0);
      long col  = c0 + wn * 32 + nt * 16 + (lane & 15);
#pragma unroll
      for (int j = 0; j < 8; ++j) {
        float v   = acc[mt][nt][j];
        long  idx = (rowb + j) * (long)N + col;
        if (mode == 0) {
          outB[idx] = (__bf16)v;
        } else if (mode == 1) {
          outF[idx] = v + bias[col] + resid[idx];
        } else {
          v += bias[col];
          outB[idx] = (__bf16)fmaxf(v, 0.0f);
        }
      }
    }
}

// ---------------------------------------------------------------------------
// Per-node 16-head attention. One wave per node (8 nodes / 256-thread block).
// ---------------------------------------------------------------------------
__global__ __launch_bounds__(256)
void k_attention(const __bf16* __restrict__ Q, const __bf16* __restrict__ K,
                 const __bf16* __restrict__ V, __bf16* __restrict__ O) {
  __shared__ __bf16 sh[8 * 3072];   // 48 KB: per-wave Q|K|V slices
  const int lane = threadIdx.x & 31;
  const int wave = threadIdx.x >> 5;
  const long node = (long)blockIdx.x * 8 + wave;

  __bf16* sq = &sh[wave * 3072];
  __bf16* sk = sq + 1024;
  __bf16* sv = sq + 2048;
  const __bf16* gq = Q + node * 1024;
  const __bf16* gk = K + node * 1024;
  const __bf16* gv = V + node * 1024;
#pragma unroll
  for (int i = 0; i < 4; ++i) {
    int c = lane + i * 32;
    ((bf16x8*)sq)[c] = ((const bf16x8*)gq)[c];
    ((bf16x8*)sk)[c] = ((const bf16x8*)gk)[c];
    ((bf16x8*)sv)[c] = ((const bf16x8*)gv)[c];
  }
  __syncthreads();

  const int h  = lane >> 1;          // head owned by lane pair
  const int gb = (lane & 1) * 8;     // this lane's 8 key-heads

  float qr[64];
#pragma unroll
  for (int d = 0; d < 64; ++d) qr[d] = (float)sq[h * 64 + d];

  float p[8];
  float mx = -3.0e38f;
#pragma unroll
  for (int j = 0; j < 8; ++j) {
    const __bf16* kr = sk + (gb + j) * 64;
    float s = 0.0f;
#pragma unroll
    for (int d = 0; d < 64; ++d) s += qr[d] * (float)kr[d];
    s *= 0.125f;                     // 1/sqrt(64)
    p[j] = s;
    mx = fmaxf(mx, s);
  }
  mx = fmaxf(mx, __shfl_xor(mx, 1));
  float sum = 0.0f;
#pragma unroll
  for (int j = 0; j < 8; ++j) { p[j] = expf(p[j] - mx); sum += p[j]; }
  sum += __shfl_xor(sum, 1);
  const float inv = 1.0f / sum;
#pragma unroll
  for (int j = 0; j < 8; ++j) p[j] *= inv;

  __bf16* out = O + node * 1024 + h * 64;
#pragma unroll
  for (int dc = 0; dc < 4; ++dc) {
    float accv[16] = {};
#pragma unroll
    for (int j = 0; j < 8; ++j) {
      const __bf16* vr = sv + (gb + j) * 64 + dc * 16;
      const float pj = p[j];
#pragma unroll
      for (int dd = 0; dd < 16; ++dd) accv[dd] += pj * (float)vr[dd];
    }
#pragma unroll
    for (int dd = 0; dd < 16; ++dd) {
      float t = accv[dd] + __shfl_xor(accv[dd], 1);
      if (!(lane & 1)) out[dc * 16 + dd] = (__bf16)t;
    }
  }
}

// ---------------------------------------------------------------------------
// Column sums / sums-of-squares for batch norm (stats pre-zeroed).
// ---------------------------------------------------------------------------
__global__ __launch_bounds__(256)
void k_colstats(const float* __restrict__ x, float* __restrict__ stats,
                int C, int rowsPerBlock) {
  const int c = threadIdx.x * 4;
  const long r0 = (long)blockIdx.x * rowsPerBlock;
  f32x4 s = {0, 0, 0, 0}, q = {0, 0, 0, 0};
  for (int r = 0; r < rowsPerBlock; ++r) {
    f32x4 v = *(const f32x4*)(x + (r0 + r) * C + c);
    s += v;
    q += v * v;
  }
  atomicAdd(&stats[c + 0], s.x);
  atomicAdd(&stats[c + 1], s.y);
  atomicAdd(&stats[c + 2], s.z);
  atomicAdd(&stats[c + 3], s.w);
  atomicAdd(&stats[C + c + 0], q.x);
  atomicAdd(&stats[C + c + 1], q.y);
  atomicAdd(&stats[C + c + 2], q.z);
  atomicAdd(&stats[C + c + 3], q.w);
}

// ---------------------------------------------------------------------------
// Batch-norm finalize + apply: y = (x-mean)*gamma*rsqrt(var+eps)+beta.
// ---------------------------------------------------------------------------
__global__ __launch_bounds__(256)
void k_bn_apply(const float* __restrict__ x, const float* __restrict__ stats,
                const float* __restrict__ gamma, const float* __restrict__ beta,
                float* __restrict__ y, __bf16* __restrict__ ybf,
                long total, int C, float invM) {
  long i = ((long)blockIdx.x * 256 + threadIdx.x) * 4;
  if (i >= total) return;
  int c = (int)(i % C);
  f32x4 v = *(const f32x4*)(x + i);
  f32x4 o;
#pragma unroll
  for (int j = 0; j < 4; ++j) {
    float mean = stats[c + j] * invM;
    float var  = stats[C + c + j] * invM - mean * mean;
    float sc   = gamma[c + j] * rsqrtf(var + BN_EPS);
    float shf  = beta[c + j] - mean * sc;
    o[j] = v[j] * sc + shf;
  }
  *(f32x4*)(y + i) = o;
  if (ybf) {
#pragma unroll
    for (int j = 0; j < 4; ++j) ybf[i + j] = (__bf16)o[j];
  }
}

// ---------------------------------------------------------------------------
// Host-side orchestration
// ---------------------------------------------------------------------------
extern "C" void kernel_launch(void* const* d_in, const int* in_sizes, int n_in,
                              void* d_out, int out_size, void* d_ws, size_t ws_size,
                              hipStream_t stream) {
  const float* h   = (const float*)d_in[0];
  const float* Wq  = (const float*)d_in[1];
  const float* Wk  = (const float*)d_in[2];
  const float* Wv  = (const float*)d_in[3];
  const float* Wo  = (const float*)d_in[4];
  const float* bo  = (const float*)d_in[5];
  const float* g1  = (const float*)d_in[6];
  const float* b1  = (const float*)d_in[7];
  const float* W1  = (const float*)d_in[8];
  const float* bf1 = (const float*)d_in[9];
  const float* W2  = (const float*)d_in[10];
  const float* bf2 = (const float*)d_in[11];
  const float* g2  = (const float*)d_in[12];
  const float* b2  = (const float*)d_in[13];
  float* out = (float*)d_out;

  const size_t MB = 1024ull * 1024ull;
  char* w = (char*)d_ws;
  __bf16* h_bf    = (__bf16*)(w + 0);          // 128 MB
  __bf16* q_bf    = (__bf16*)(w + 128 * MB);   // 128 MB (reused: x1n_bf)
  __bf16* k_bf    = (__bf16*)(w + 256 * MB);   // 128 MB (reused: ffh lower)
  __bf16* v_bf    = (__bf16*)(w + 384 * MB);   // 128 MB (reused: ffh upper)
  __bf16* attn_bf = (__bf16*)(w + 512 * MB);   // 128 MB
  float*  x1      = (float*)(w + 640 * MB);    // 256 MB
  float*  x1n     = (float*)(w + 896 * MB);    // 256 MB
  __bf16* WqT     = (__bf16*)(w + 1152 * MB);
  __bf16* WkT     = (__bf16*)(w + 1154 * MB);
  __bf16* WvT     = (__bf16*)(w + 1156 * MB);
  __bf16* WoT     = (__bf16*)(w + 1158 * MB);
  __bf16* W1T     = (__bf16*)(w + 1160 * MB);  // 4 MB [2048,1024]
  __bf16* W2T     = (__bf16*)(w + 1164 * MB);  // 4 MB [1024,2048]
  float*  stats1  = (float*)(w + 1168 * MB);
  float*  stats2  = stats1 + 2 * DMODEL;
  __bf16* x1n_bf  = q_bf;                      // reuse (Q consumed)
  __bf16* ffh     = k_bf;                      // reuse 256 MB span (K+V consumed)

  const size_t needed = 1168 * MB + 32 * 1024;
  if (ws_size < needed) return;

  const long totalHD = (long)N_NODES * DMODEL;

  (void)hipMemsetAsync(stats1, 0, 4 * DMODEL * sizeof(float), stream);

  // 1) conversions
  k_f32_to_bf16<<<65536, 256, 0, stream>>>(h, h_bf, totalHD);
  k_transpose_bf16<<<4096, 256, 0, stream>>>(Wq, WqT, DMODEL, DMODEL);
  k_transpose_bf16<<<4096, 256, 0, stream>>>(Wk, WkT, DMODEL, DMODEL);
  k_transpose_bf16<<<4096, 256, 0, stream>>>(Wv, WvT, DMODEL, DMODEL);
  k_transpose_bf16<<<4096, 256, 0, stream>>>(Wo, WoT, DMODEL, DMODEL);
  k_transpose_bf16<<<8192, 256, 0, stream>>>(W1, W1T, DMODEL, FFDIM);
  k_transpose_bf16<<<8192, 256, 0, stream>>>(W2, W2T, FFDIM, DMODEL);

  dim3 gD(N_NODES / 128, DMODEL / 128);  // (512, 8)
  dim3 gF(N_NODES / 128, FFDIM / 128);   // (512, 16)

  // 2) Q/K/V projections (bf16 out)
  k_gemm_bf16<<<gD, 256, 0, stream>>>(h_bf, WqT, N_NODES, DMODEL, DMODEL, 0,
                                      nullptr, nullptr, nullptr, q_bf);
  k_gemm_bf16<<<gD, 256, 0, stream>>>(h_bf, WkT, N_NODES, DMODEL, DMODEL, 0,
                                      nullptr, nullptr, nullptr, k_bf);
  k_gemm_bf16<<<gD, 256, 0, stream>>>(h_bf, WvT, N_NODES, DMODEL, DMODEL, 0,
                                      nullptr, nullptr, nullptr, v_bf);

  // 3) per-node multi-head attention
  k_attention<<<N_NODES / 8, 256, 0, stream>>>(q_bf, k_bf, v_bf, attn_bf);

  // 4) output projection + bias + residual -> x1 (fp32)
  k_gemm_bf16<<<gD, 256, 0, stream>>>(attn_bf, WoT, N_NODES, DMODEL, DMODEL, 1,
                                      bo, h, x1, nullptr);

  // 5) batch norm 1 -> x1n (fp32) + x1n_bf (bf16)
  k_colstats<<<256, 256, 0, stream>>>(x1, stats1, DMODEL, N_NODES / 256);
  k_bn_apply<<<65536, 256, 0, stream>>>(x1, stats1, g1, b1, x1n, x1n_bf,
                                        totalHD, DMODEL, 1.0f / N_NODES);

  // 6) FFN
  k_gemm_bf16<<<gF, 256, 0, stream>>>(x1n_bf, W1T, N_NODES, FFDIM, DMODEL, 2,
                                      bf1, nullptr, nullptr, ffh);
  k_gemm_bf16<<<gD, 256, 0, stream>>>(ffh, W2T, N_NODES, DMODEL, FFDIM, 1,
                                      bf2, x1n, out, nullptr);

  // 7) batch norm 2 (in place on d_out)
  k_colstats<<<256, 256, 0, stream>>>(out, stats2, DMODEL, N_NODES / 256);
  k_bn_apply<<<65536, 256, 0, stream>>>(out, stats2, g2, b2, out, nullptr,
                                        totalHD, DMODEL, 1.0f / N_NODES);
}
